// Mamba_model_3212635538148
// MI455X (gfx1250) — matrix operand order repaired and checked by reading, not compiled, not run on hardware
//
#include <hip/hip_runtime.h>
#include <hip/hip_bf16.h>
#include <math.h>

typedef __attribute__((ext_vector_type(16))) _Float16 v16h;
typedef __attribute__((ext_vector_type(8)))  _Float16 v8h;
typedef __attribute__((ext_vector_type(4)))  _Float16 v4h;
typedef __attribute__((ext_vector_type(8)))  float    v8f;
typedef int v4i_vec __attribute__((vector_size(4 * sizeof(int))));

#define B_     16
#define L_     512
#define ENC    7
#define DM     512
#define DIN    1024
#define DSTATE 16
#define DCONV  4
#define DTRANK 32
#define COUT   7
#define PRED   96
#define MTOT   (B_ * L_)   // 8192 rows for all GEMMs

#if defined(__gfx1250__) && __has_builtin(__builtin_amdgcn_global_load_async_to_lds_b128)
#define HAVE_ASYNC_LDS 1
#else
#define HAVE_ASYNC_LDS 0
#endif

#if HAVE_ASYNC_LDS
// Builtin signature (from clang diagnostic): param0 is v4i in AS(1) ("__device__").
#define ASYNC_LDS_B128(gsrc, ldst)                                              \
    __builtin_amdgcn_global_load_async_to_lds_b128(                             \
        (__attribute__((address_space(1))) v4i_vec*)(gsrc),                     \
        (__attribute__((address_space(3))) v4i_vec*)(ldst), 0, 0)
#endif

__device__ __forceinline__ void wait_asynccnt0() {
#if defined(__gfx1250__)
#if __has_builtin(__builtin_amdgcn_s_wait_asynccnt)
    __builtin_amdgcn_s_wait_asynccnt(0);
#else
    asm volatile("s_wait_asynccnt 0" ::: "memory");
#endif
#endif
}

// ---------------------------------------------------------------------------
// Per-(b,c) mean / std over L (biased variance, matching jnp.var)
// ---------------------------------------------------------------------------
__global__ void stats_kernel(const float* __restrict__ x_enc, float* __restrict__ stats) {
    int t = blockIdx.x * blockDim.x + threadIdx.x;
    if (t >= B_ * ENC) return;
    int b = t / ENC, c = t % ENC;
    float s = 0.f, s2 = 0.f;
    for (int l = 0; l < L_; ++l) {
        float v = x_enc[(b * L_ + l) * ENC + c];
        s += v; s2 += v * v;
    }
    float mean = s / (float)L_;
    float var  = s2 / (float)L_ - mean * mean;
    stats[t]            = mean;
    stats[B_ * ENC + t] = sqrtf(var + 1e-5f);
}

// ---------------------------------------------------------------------------
// Normalize + wrap-padded 3-tap embedding + sinusoidal pos-emb -> x (f16)
// ---------------------------------------------------------------------------
__global__ void embed_kernel(const float* __restrict__ x_enc,
                             const float* __restrict__ W_emb,
                             const float* __restrict__ stats,
                             _Float16* __restrict__ x_h) {
    int idx = blockIdx.x * blockDim.x + threadIdx.x;
    if (idx >= MTOT * DM) return;
    int d = idx % DM;
    int l = (idx / DM) % L_;
    int b = idx / (DM * L_);
    float tok = 0.f;
    #pragma unroll
    for (int k = 0; k < 3; ++k) {
        int ls = (l + k - 1 + L_) % L_;       // wrap padding
        #pragma unroll
        for (int c = 0; c < ENC; ++c) {
            float mean = stats[b * ENC + c];
            float std  = stats[B_ * ENC + b * ENC + c];
            float xn   = (x_enc[(b * L_ + ls) * ENC + c] - mean) / std;
            tok += xn * W_emb[(k * ENC + c) * DM + d];
        }
    }
    int   i2   = (d >> 1) * 2;
    float freq = __expf(-(logf(10000.f) / (float)DM) * (float)i2);
    float ang  = (float)l * freq;
    float pe   = (d & 1) ? __cosf(ang) : __sinf(ang);
    x_h[idx] = (_Float16)(tok + pe);
}

// ---------------------------------------------------------------------------
// f32 -> f16 weight conversion
// ---------------------------------------------------------------------------
__global__ void cvt_kernel(const float* __restrict__ in, _Float16* __restrict__ out, int n) {
    int i = blockIdx.x * blockDim.x + threadIdx.x;
    if (i < n) out[i] = (_Float16)in[i];
}

// ---------------------------------------------------------------------------
// WMMA GEMM: C(f16) = A(f16, MxK row-major) * B(f16, KxN row-major), f32 accum
// Block tile 128(M) x 32(N), 8 waves; each wave: 16x32 output = 2 accumulators,
// one shared A fragment -> 2 v_wmma per 32-K step.
// B slab (32K x 32N) staged transposed in LDS: b64 global load + 4 ds_store_b16
// per thread; per-lane fragment reads are contiguous ds_load_b128 pairs.
// grid.x = M/128, grid.y = N/32, block = 256
// ---------------------------------------------------------------------------
__global__ void wmma_gemm_f16(const _Float16* __restrict__ A, int lda,
                              const _Float16* __restrict__ Bm, int ldb,
                              _Float16* __restrict__ C, int ldc, int K) {
    __shared__ _Float16 Bs[32 * 32];          // [n (32)][k (32)] transposed slab
    const int n0   = blockIdx.y * 32;
    const int m0   = blockIdx.x * 128;
    const int wave = threadIdx.x >> 5;
    const int lane = threadIdx.x & 31;
    const int half = lane >> 4;
    const int lm   = lane & 15;
    const int m    = m0 + wave * 16 + lm;

    const int sr = threadIdx.x & 31;          // staging: K row
    const int sc = (threadIdx.x >> 5) * 4;    // staging: N col base (0..28)

    v8f acc0 = {}, acc1 = {};
    for (int k0 = 0; k0 < K; k0 += 32) {
        // Stage B[k0..k0+31, n0..n0+31] transposed: 1024 halfs, 4 per thread.
        {
            v4h bv = *(const v4h*)&Bm[(size_t)(k0 + sr) * ldb + n0 + sc];
            #pragma unroll
            for (int j = 0; j < 4; ++j)
                Bs[(sc + j) * 32 + sr] = bv[j];
        }
        __syncthreads();
        if (k0 + 32 < K)
            __builtin_prefetch(&Bm[(size_t)(k0 + 32) * ldb + n0], 0, 3);

        // A fragment: lane half h holds K = {h*8..h*8+7} then {16+h*8..+7}
        const _Float16* arow = A + (size_t)m * lda + k0 + half * 8;
        v8h a_lo = *(const v8h*)(arow);
        v8h a_hi = *(const v8h*)(arow + 16);
        // B fragments: columns lm and 16+lm; lane half h holds K = {h*8..h*8+7} then {16+h*8..+7}
        const _Float16* b0 = &Bs[lm * 32 + half * 8];
        const _Float16* b1 = &Bs[(16 + lm) * 32 + half * 8];
        v8h b0_lo = *(const v8h*)(b0), b0_hi = *(const v8h*)(b0 + 16);
        v8h b1_lo = *(const v8h*)(b1), b1_hi = *(const v8h*)(b1 + 16);

        v16h a, bf0, bf1;
        #pragma unroll
        for (int e = 0; e < 8; ++e) {
            a[e]   = a_lo[e];  a[8 + e]   = a_hi[e];
            bf0[e] = b0_lo[e]; bf0[8 + e] = b0_hi[e];
            bf1[e] = b1_lo[e]; bf1[8 + e] = b1_hi[e];
        }
        acc0 = __builtin_amdgcn_wmma_f32_16x16x32_f16(
                   false, a, false, bf0, (short)0, acc0, false, false);
        acc1 = __builtin_amdgcn_wmma_f32_16x16x32_f16(
                   false, a, false, bf1, (short)0, acc1, false, false);
        __syncthreads();
    }
    // C/D layout: VGPR r -> row (half*8 + r), lane lm -> column
    #pragma unroll
    for (int r = 0; r < 8; ++r) {
        int row = m0 + wave * 16 + half * 8 + r;
        C[(size_t)row * ldc + n0 + lm]      = (_Float16)acc0[r];
        C[(size_t)row * ldc + n0 + 16 + lm] = (_Float16)acc1[r];
    }
}

// ---------------------------------------------------------------------------
// Causal depthwise conv (4 taps) + bias + SiLU on xm = xz[..., :DIN]
// ---------------------------------------------------------------------------
__global__ void conv_silu_kernel(const _Float16* __restrict__ xz,
                                 const float* __restrict__ conv_w,
                                 const float* __restrict__ conv_b,
                                 _Float16* __restrict__ u) {
    int idx = blockIdx.x * blockDim.x + threadIdx.x;
    if (idx >= MTOT * DIN) return;
    int d  = idx % DIN;
    int bl = idx / DIN;
    int l  = bl % L_;
    int b  = bl / L_;
    float acc = conv_b[d];
    #pragma unroll
    for (int k = 0; k < DCONV; ++k) {
        int ls = l + k - (DCONV - 1);
        if (ls >= 0)
            acc += (float)xz[(size_t)(b * L_ + ls) * (2 * DIN) + d] * conv_w[k * DIN + d];
    }
    float s = acc / (1.f + __expf(-acc));   // silu
    u[idx] = (_Float16)s;
}

// ---------------------------------------------------------------------------
// In-place dt = softplus(dt_pre + b_dt)
// ---------------------------------------------------------------------------
__global__ void softplus_kernel(_Float16* __restrict__ dt, const float* __restrict__ b_dt, int n) {
    int i = blockIdx.x * blockDim.x + threadIdx.x;
    if (i >= n) return;
    float v  = (float)dt[i] + b_dt[i % DIN];
    float sp = (v > 20.f) ? v : log1pf(__expf(v));
    dt[i] = (_Float16)sp;
}

// ---------------------------------------------------------------------------
// Selective scan, fused with D-skip and silu(z) gating.
// One thread per (b, d); 16-wide state in registers.
// Per-timestep B/C slab (64 B) is double-buffered in LDS and prefetched with
// GLOBAL_LOAD_ASYNC_TO_LDS_B128 (ASYNCcnt) one step ahead of the VALU compute.
// grid = B_*(DIN/256), block = 256
// ---------------------------------------------------------------------------
__global__ void scan_kernel(const _Float16* __restrict__ dt,
                            const _Float16* __restrict__ u,
                            const _Float16* __restrict__ dbc,
                            const _Float16* __restrict__ xz,
                            const float* __restrict__ A_log,
                            const float* __restrict__ Dp,
                            _Float16* __restrict__ y) {
    const int b = blockIdx.x / (DIN / 256);
    const int d = (blockIdx.x % (DIN / 256)) * 256 + threadIdx.x;
    __shared__ _Float16 sBC[2][2 * DSTATE];   // [buf][B(16) | C(16)]

    float Arow[DSTATE], h[DSTATE];
    const float Dd = Dp[d];
    #pragma unroll
    for (int n = 0; n < DSTATE; ++n) {
        Arow[n] = -__expf(A_log[d * DSTATE + n]);
        h[n] = 0.f;
    }

    const size_t dbcStride = DTRANK + 2 * DSTATE;   // 64
    const _Float16* dbc_b  = dbc + (size_t)(b * L_) * dbcStride + DTRANK;

    // Prologue: stage slab for t = 0 into buffer 0.
#if HAVE_ASYNC_LDS
    if (threadIdx.x < 4)
        ASYNC_LDS_B128(dbc_b + threadIdx.x * 8, &sBC[0][threadIdx.x * 8]);
    wait_asynccnt0();
#else
    if (threadIdx.x < 2 * DSTATE)
        sBC[0][threadIdx.x] = dbc_b[threadIdx.x];
#endif
    __syncthreads();

    for (int t = 0; t < L_; ++t) {
        const int cur = t & 1, nxt = cur ^ 1;
        // Async prefetch of next timestep's B/C slab while computing this one.
        if (t + 1 < L_) {
#if HAVE_ASYNC_LDS
            if (threadIdx.x < 4)
                ASYNC_LDS_B128(dbc_b + (size_t)(t + 1) * dbcStride + threadIdx.x * 8,
                               &sBC[nxt][threadIdx.x * 8]);
#else
            if (threadIdx.x < 2 * DSTATE)
                sBC[nxt][threadIdx.x] = dbc_b[(size_t)(t + 1) * dbcStride + threadIdx.x];
#endif
        }

        size_t row = (size_t)(b * L_ + t) * DIN + d;
        float dtv = (float)dt[row];
        float uv  = (float)u[row];
        float du  = dtv * uv;
        float yv  = 0.f;
        #pragma unroll
        for (int n = 0; n < DSTATE; ++n) {
            float dA = __expf(dtv * Arow[n]);
            h[n] = dA * h[n] + du * (float)sBC[cur][n];
            yv  += h[n] * (float)sBC[cur][DSTATE + n];
        }
        yv += uv * Dd;                                  // D skip
        float zv = (float)xz[(size_t)(b * L_ + t) * (2 * DIN) + DIN + d];
        yv *= zv / (1.f + __expf(-zv));                 // * silu(z)
        y[row] = (_Float16)yv;

#if HAVE_ASYNC_LDS
        wait_asynccnt0();
#endif
        __syncthreads();
    }
}

// ---------------------------------------------------------------------------
// Head: out = (out1 @ W_head) * std + mean, last PRED steps only
// ---------------------------------------------------------------------------
__global__ void head_kernel(const _Float16* __restrict__ out1,
                            const float* __restrict__ W_head,
                            const float* __restrict__ stats,
                            float* __restrict__ out) {
    int t = blockIdx.x * blockDim.x + threadIdx.x;
    if (t >= B_ * PRED * COUT) return;
    int c = t % COUT;
    int r = (t / COUT) % PRED;
    int b = t / (COUT * PRED);
    int l = L_ - PRED + r;
    float acc = 0.f;
    for (int dd = 0; dd < DM; ++dd)
        acc += (float)out1[(size_t)(b * L_ + l) * DM + dd] * W_head[dd * COUT + c];
    float mean = stats[b * ENC + c];
    float std  = stats[B_ * ENC + b * ENC + c];
    out[t] = acc * std + mean;
}

// ---------------------------------------------------------------------------
extern "C" void kernel_launch(void* const* d_in, const int* in_sizes, int n_in,
                              void* d_out, int out_size, void* d_ws, size_t ws_size,
                              hipStream_t stream) {
    (void)in_sizes; (void)n_in; (void)out_size; (void)ws_size;
    const float* x_enc   = (const float*)d_in[0];
    const float* W_emb   = (const float*)d_in[1];
    const float* W_in    = (const float*)d_in[2];
    const float* conv_w  = (const float*)d_in[3];
    const float* conv_b  = (const float*)d_in[4];
    const float* W_xproj = (const float*)d_in[5];
    const float* W_dt    = (const float*)d_in[6];
    const float* b_dt    = (const float*)d_in[7];
    const float* A_log   = (const float*)d_in[8];
    const float* Dp      = (const float*)d_in[9];
    const float* W_out   = (const float*)d_in[10];
    const float* W_head  = (const float*)d_in[11];
    float* out = (float*)d_out;

    char*  ws  = (char*)d_ws;
    size_t off = 0;
    auto alloc = [&](size_t bytes) -> char* {
        char* p = ws + off;
        off += (bytes + 255) & ~(size_t)255;
        return p;
    };
    float*    stats  = (float*)   alloc((size_t)2 * B_ * ENC * sizeof(float));
    _Float16* x_h    = (_Float16*)alloc((size_t)MTOT * DM * 2);
    _Float16* Win_h  = (_Float16*)alloc((size_t)DM * 2 * DIN * 2);
    _Float16* Wxp_h  = (_Float16*)alloc((size_t)DIN * (DTRANK + 2 * DSTATE) * 2);
    _Float16* Wdt_h  = (_Float16*)alloc((size_t)DTRANK * DIN * 2);
    _Float16* Wout_h = (_Float16*)alloc((size_t)DIN * DM * 2);
    _Float16* xz_h   = (_Float16*)alloc((size_t)MTOT * 2 * DIN * 2);
    _Float16* u_h    = (_Float16*)alloc((size_t)MTOT * DIN * 2);
    _Float16* dbc_h  = (_Float16*)alloc((size_t)MTOT * (DTRANK + 2 * DSTATE) * 2);
    _Float16* dt_h   = (_Float16*)alloc((size_t)MTOT * DIN * 2);
    _Float16* y_h    = (_Float16*)alloc((size_t)MTOT * DIN * 2);
    _Float16* out1_h = (_Float16*)alloc((size_t)MTOT * DM * 2);

    stats_kernel<<<1, 128, 0, stream>>>(x_enc, stats);
    embed_kernel<<<(MTOT * DM + 255) / 256, 256, 0, stream>>>(x_enc, W_emb, stats, x_h);

    cvt_kernel<<<(DM * 2 * DIN + 255) / 256, 256, 0, stream>>>(W_in, Win_h, DM * 2 * DIN);
    cvt_kernel<<<(DIN * 64 + 255) / 256, 256, 0, stream>>>(W_xproj, Wxp_h, DIN * 64);
    cvt_kernel<<<(DTRANK * DIN + 255) / 256, 256, 0, stream>>>(W_dt, Wdt_h, DTRANK * DIN);
    cvt_kernel<<<(DIN * DM + 255) / 256, 256, 0, stream>>>(W_out, Wout_h, DIN * DM);

    // xz = x @ W_in            (8192 x 512 x 2048)
    wmma_gemm_f16<<<dim3(MTOT / 128, (2 * DIN) / 32), 256, 0, stream>>>(
        x_h, DM, Win_h, 2 * DIN, xz_h, 2 * DIN, DM);

    conv_silu_kernel<<<(MTOT * DIN + 255) / 256, 256, 0, stream>>>(xz_h, conv_w, conv_b, u_h);

    // dbc = u @ W_xproj        (8192 x 1024 x 64)
    wmma_gemm_f16<<<dim3(MTOT / 128, 64 / 32), 256, 0, stream>>>(
        u_h, DIN, Wxp_h, 64, dbc_h, 64, DIN);

    // dt_pre = dbc[:, :32] @ W_dt   (8192 x 32 x 1024), lda = 64 picks first 32 cols
    wmma_gemm_f16<<<dim3(MTOT / 128, DIN / 32), 256, 0, stream>>>(
        dbc_h, 64, Wdt_h, DIN, dt_h, DIN, DTRANK);

    softplus_kernel<<<(MTOT * DIN + 255) / 256, 256, 0, stream>>>(dt_h, b_dt, MTOT * DIN);

    scan_kernel<<<B_ * (DIN / 256), 256, 0, stream>>>(dt_h, u_h, dbc_h, xz_h, A_log, Dp, y_h);

    // out1 = y @ W_out         (8192 x 1024 x 512)
    wmma_gemm_f16<<<dim3(MTOT / 128, DM / 32), 256, 0, stream>>>(
        y_h, DIN, Wout_h, DM, out1_h, DM, DIN);

    head_kernel<<<(B_ * PRED * COUT + 255) / 256, 256, 0, stream>>>(out1_h, W_head, stats, out);
}
